// DeformTransformerBlock1D_82437602280006
// MI455X (gfx1250) — matrix operand order repaired and checked by reading, not compiled, not run on hardware
//
#include <hip/hip_runtime.h>
#include <hip/hip_bf16.h>
#include <math.h>

// ---------------------------------------------------------------------------
// DeformTransformerBlock1D for MI455X (gfx1250, wave32, WMMA f16 16x16x32).
// Roofline: ~26 GFLOP vs ~300MB traffic @23.3TB/s (~13us) -> f16 WMMA
// (~10us compute) balances; f32 WMMA (16x16x4) would be ~8x slower.
// v-row staging to LDS uses the Tensor Data Mover (tensor_load_to_lds).
// ---------------------------------------------------------------------------

typedef __attribute__((ext_vector_type(16))) _Float16     v16h;
typedef __attribute__((ext_vector_type(8)))  float        v8f;
typedef __attribute__((ext_vector_type(4)))  unsigned int u32x4;
typedef __attribute__((ext_vector_type(8)))  int          i32x8;
typedef __attribute__((ext_vector_type(4)))  int          i32x4;

#define CC    128
#define GG    8
#define PP    4
#define HHH   96
#define WWID  320
#define BBB   4
#define HW    (HHH * WWID)          // 30720
#define NTOK  (BBB * HW)            // 122880
#define FOUT  (2 * CC * HW)         // 7864320 floats per feature-map output
#define KPOFF (2 * FOUT)            // key_points offset in d_out

// ---- workspace layout ------------------------------------------------------
// weights repacked to f16 WMMA-B operand order (offsets in f16 units):
#define OFF_WV    0        // 128x128 -> 16384
#define OFF_WOFF  16384    // 128x32  ->  4096
#define OFF_WATTN 20480    // 128x32  ->  4096
#define OFF_WOUT  24576    // 128x128 -> 16384
#define OFF_W1    40960    // 128x256 -> 32768
#define OFF_W2    73728    // 256x128 -> 32768
#define NWF16     106496
#define WSB_V     262144                        // f16 [NTOK][128]
#define WSB_ATTN  (WSB_V    + NTOK * CC * 2)    // f32 [NTOK][32]
#define WSB_POS   (WSB_ATTN + NTOK * 32 * 4)    // f32 [NTOK][32]
#define WSB_XLN   (WSB_POS  + NTOK * 32 * 4)    // f32 [NTOK][128]

// A-operand address for 16-bit 16x32 fragments chained over K (32/k-step).
// ISA 7.12.2: lane = 16*((k%16)/8) + m ; elem = 8*(k/16 within step) + (k%8).
__device__ __forceinline__ int a_frag_addr(int m, int k) {
  int s    = k >> 5;
  int kk   = k & 31;
  int half = (kk & 15) >> 3;
  int e    = ((kk >> 4) << 3) + (kk & 7);
  return s * 512 + (half * 16 + m) * 16 + e;
}

// ---------------------------------------------------------------------------
// Kernel 0: repack weights into f16 WMMA-B operand order.
// B 32x16 layout: lane = 16*((k%16)/8) + n ; elem = 8*(k/16) + (k%8) (matches A order)
// ---------------------------------------------------------------------------
__global__ void __launch_bounds__(256) kprep(
    const float* __restrict__ Wv,   const float* __restrict__ Woff,
    const float* __restrict__ Wattn,const float* __restrict__ Wout,
    const float* __restrict__ W1,   const float* __restrict__ W2,
    _Float16* __restrict__ wdst)
{
  int idx = blockIdx.x * 256 + threadIdx.x;
  if (idx >= NWF16) return;
  const float* src; int base, Ks, Nn, Kk; bool kn;
  if      (idx < 16384) { src = Wv;    base = OFF_WV;    Ks = 4; Nn = 128; Kk = 128; kn = true;  }
  else if (idx < 20480) { src = Woff;  base = OFF_WOFF;  Ks = 4; Nn = 32;  Kk = 128; kn = true;  }
  else if (idx < 24576) { src = Wattn; base = OFF_WATTN; Ks = 4; Nn = 32;  Kk = 128; kn = true;  }
  else if (idx < 40960) { src = Wout;  base = OFF_WOUT;  Ks = 4; Nn = 128; Kk = 128; kn = true;  }
  else if (idx < 73728) { src = W1;    base = OFF_W1;    Ks = 4; Nn = 256; Kk = 128; kn = false; }
  else                  { src = W2;    base = OFF_W2;    Ks = 8; Nn = 128; Kk = 256; kn = false; }
  int local = idx - base;
  int e    = local & 15;
  int lane = (local >> 4) & 31;
  int rest = local >> 9;
  int s    = rest % Ks;
  int nt   = rest / Ks;
  int nn   = lane & 15;
  int kk   = ((e >> 3) << 4) + ((lane >> 4) << 3) + (e & 7);
  int k    = s * 32 + kk;
  int n    = nt * 16 + nn;
  float v  = kn ? src[k * Nn + n] : src[n * Kk + k];
  wdst[idx] = (_Float16)v;
}

// ---------------------------------------------------------------------------
// Kernel 1: projections. Per wave: 16-token tile.
//   v = kv @ W_v + b_v (f16 out), off/attn = q @ W + b, softmax(P=4),
//   pos = clip(anchor*319 + off), key_points -> d_out, attn/pos -> ws.
// ---------------------------------------------------------------------------
__global__ void __launch_bounds__(256) kproj(
    const float* __restrict__ f1, const float* __restrict__ f2,
    const float* __restrict__ anchor,
    const float* __restrict__ b_v, const float* __restrict__ b_off,
    const float* __restrict__ b_attn,
    const _Float16* __restrict__ wb,
    _Float16* __restrict__ ws_v, float* __restrict__ ws_attn,
    float* __restrict__ ws_pos, float* __restrict__ d_out)
{
  extern __shared__ char smem[];
  const int wave = threadIdx.x >> 5;
  const int lane = threadIdx.x & 31;
  _Float16* Akv = (_Float16*)smem + wave * 2048;
  _Float16* Aq  = (_Float16*)(smem + 32768) + wave * 2048;
  float* offb   = (float*)(smem + 65536) + wave * 512;
  float* attnb  = (float*)(smem + 81920) + wave * 512;

  const int blockTok0 = blockIdx.x * 128;
  const int tok0 = blockTok0 + wave * 16;   // 16 | 320 -> tile stays in one row
  const int b   = tok0 / HW;
  const int rem = tok0 % HW;
  const int h   = rem / WWID;
  const int w0  = rem % WWID;
  const float* qsrc  = (b < 2) ? (f1 + b * CC * HW) : (f2 + (b - 2) * CC * HW);
  const float* kvsrc = (b < 2) ? (f2 + b * CC * HW) : (f1 + (b - 2) * CC * HW);

  for (int i = lane; i < 2048; i += 32) {
    int c = i >> 4, m = i & 15;
    int g = c * HW + h * WWID + w0 + m;
    int a = a_frag_addr(m, c);
    Aq[a]  = (_Float16)qsrc[g];
    Akv[a] = (_Float16)kvsrc[g];
  }
  __syncthreads();

  v16h aq[4], akv[4];
#pragma unroll
  for (int s = 0; s < 4; ++s) {
    aq[s]  = *(const v16h*)(Aq  + s * 512 + lane * 16);
    akv[s] = *(const v16h*)(Akv + s * 512 + lane * 16);
  }
  const int nn = lane & 15;
  const int mb = (lane >> 4) * 8;

  // v = kv @ W_v
#pragma unroll
  for (int nt = 0; nt < 8; ++nt) {
    v8f acc = {};
#pragma unroll
    for (int s = 0; s < 4; ++s) {
      v16h bf = *(const v16h*)(wb + OFF_WV + (nt * 4 + s) * 512 + lane * 16);
      acc = __builtin_amdgcn_wmma_f32_16x16x32_f16(false, akv[s], false, bf,
                                                   (short)0, acc, false, false);
    }
    int n = nt * 16 + nn;
    float bias = b_v[n];
#pragma unroll
    for (int r = 0; r < 8; ++r)
      ws_v[(tok0 + mb + r) * CC + n] = (_Float16)(acc[r] + bias);
  }
  // off logits
#pragma unroll
  for (int nt = 0; nt < 2; ++nt) {
    v8f acc = {};
#pragma unroll
    for (int s = 0; s < 4; ++s) {
      v16h bf = *(const v16h*)(wb + OFF_WOFF + (nt * 4 + s) * 512 + lane * 16);
      acc = __builtin_amdgcn_wmma_f32_16x16x32_f16(false, aq[s], false, bf,
                                                   (short)0, acc, false, false);
    }
    int n = nt * 16 + nn;
    float bias = b_off[n];
#pragma unroll
    for (int r = 0; r < 8; ++r) offb[(mb + r) * 32 + n] = acc[r] + bias;
  }
  // attn logits
#pragma unroll
  for (int nt = 0; nt < 2; ++nt) {
    v8f acc = {};
#pragma unroll
    for (int s = 0; s < 4; ++s) {
      v16h bf = *(const v16h*)(wb + OFF_WATTN + (nt * 4 + s) * 512 + lane * 16);
      acc = __builtin_amdgcn_wmma_f32_16x16x32_f16(false, aq[s], false, bf,
                                                   (short)0, acc, false, false);
    }
    int n = nt * 16 + nn;
    float bias = b_attn[n];
#pragma unroll
    for (int r = 0; r < 8; ++r) attnb[(mb + r) * 32 + n] = acc[r] + bias;
  }
  __syncthreads();

  // softmax over P=4, positions, key_points. 128 tokens * 8 groups per block.
  for (int t = threadIdx.x; t < 1024; t += 256) {
    int tb = t >> 3, g = t & 7;
    int wv = tb >> 4, m = tb & 15;
    const float* ob = (const float*)(smem + 65536) + wv * 512 + m * 32 + g * 4;
    const float* ab = (const float*)(smem + 81920) + wv * 512 + m * 32 + g * 4;
    int tok = blockTok0 + tb;
    float anc = anchor[tok];
    float l0 = ab[0], l1 = ab[1], l2 = ab[2], l3 = ab[3];
    float mx = fmaxf(fmaxf(l0, l1), fmaxf(l2, l3));
    float e0 = __expf(l0 - mx), e1 = __expf(l1 - mx),
          e2 = __expf(l2 - mx), e3 = __expf(l3 - mx);
    float inv = 1.0f / (e0 + e1 + e2 + e3);
    float* wa = ws_attn + tok * 32 + g * 4;
    wa[0] = e0 * inv; wa[1] = e1 * inv; wa[2] = e2 * inv; wa[3] = e3 * inv;
    float* wp = ws_pos + tok * 32 + g * 4;
    float* kp = d_out + KPOFF + tok * 32 + g * 4;
#pragma unroll
    for (int p = 0; p < 4; ++p) {
      float pos = anc * 319.0f + ob[p];
      pos = fminf(fmaxf(pos, 0.0f), 319.0f);
      wp[p] = pos;
      kp[p] = pos * (1.0f / 319.0f);
    }
  }
}

// ---------------------------------------------------------------------------
// Kernel 2: one workgroup per (b,h) row. v-row (320x128 f16 = 80KB) pulled
// into LDS by the Tensor Data Mover; deformable 1-D gather+lerp; agg -> LDS
// (A-operand order); W_out GEMM; residual; LayerNorm -> ws_xln.
// LDS: 240KB of the 320KB WGP budget.
// ---------------------------------------------------------------------------
__global__ void __launch_bounds__(256) ksample(
    const float* __restrict__ f1, const float* __restrict__ f2,
    const float* __restrict__ b_out, const float* __restrict__ ln_w,
    const float* __restrict__ ln_b, const _Float16* __restrict__ wb,
    const _Float16* __restrict__ ws_v, const float* __restrict__ ws_attn,
    const float* __restrict__ ws_pos, float* __restrict__ ws_xln)
{
  extern __shared__ char smem[];
  _Float16* vbuf   = (_Float16*)smem;             // [320][128] f16 (phases 1-2)
  float*    xrow   = (float*)smem;                // [320][128] f32 (phases 3-4)
  _Float16* aggbuf = (_Float16*)(smem + 163840);  // 20 tiles, A-operand order

  const int row  = blockIdx.x;                    // b*96 + h
  const int row0 = row * WWID;
  const int b    = row / HHH;
  const int h    = row % HHH;
  const int wave = threadIdx.x >> 5;
  const int lane = threadIdx.x & 31;

  // ---- phase 1: TDM copy of v row (81920 B) into LDS -----------------------
#if __has_builtin(__builtin_amdgcn_tensor_load_to_lds)
  if (threadIdx.x < 32) {                 // one TDM issue per workgroup
    unsigned long long ga =
        (unsigned long long)(uintptr_t)(ws_v + (size_t)row0 * CC);
    unsigned int la = (unsigned int)(uintptr_t)vbuf;
    // D# group0: count=1 | lds_addr | global_addr[56:0] | type=2 (ISA 8.3)
    u32x4 g0;
    g0[0] = 1u;
    g0[1] = la;
    g0[2] = (unsigned int)ga;
    g0[3] = (unsigned int)(ga >> 32) | (2u << 30);
    // D# group1: data_size=8B; tensor_dim0=tile_dim0=10240; tile_dim1=1;
    // tensor_dim1=1; tensor_dim0_stride=10240 (ISA 8.4 bit packing)
    i32x8 g1;
    g1[0] = 3 << 16;                  // data_size
    g1[1] = (int)(10240u << 16);      // tensor_dim0[15:0] @ bits[63:48]
    g1[2] = 1 << 16;                  // tensor_dim0[31:16]=0, tensor_dim1[15:0]=1
    g1[3] = (int)(10240u << 16);      // tensor_dim1[31:16]=0, tile_dim0=10240
    g1[4] = 1;                        // tile_dim1=1, tile_dim2=0
    g1[5] = 10240;                    // tensor_dim0_stride[31:0]
    g1[6] = 0;
    g1[7] = 0;
    i32x4 z4 = {0, 0, 0, 0};
    i32x8 z8 = {0, 0, 0, 0, 0, 0, 0, 0};
    __builtin_amdgcn_tensor_load_to_lds(g0, g1, z4, z4, z8, 0);
    __builtin_amdgcn_s_wait_tensorcnt(0);
  }
#else
  {
    const uint4* src = (const uint4*)(ws_v + row0 * CC);
    uint4* dst = (uint4*)vbuf;
    for (int i = threadIdx.x; i < 5120; i += 256) dst[i] = src[i];
  }
#endif
  __syncthreads();

  // ---- phase 2: one task per (token, group): 16-channel gather+lerp --------
  for (int idx = threadIdx.x; idx < 2560; idx += 256) {
    int w = idx >> 3, g = idx & 7;
    const float* at = ws_attn + (row0 + w) * 32 + g * 4;
    const float* ps = ws_pos  + (row0 + w) * 32 + g * 4;
    float acc[16];
#pragma unroll
    for (int c = 0; c < 16; ++c) acc[c] = 0.0f;
#pragma unroll
    for (int p = 0; p < 4; ++p) {
      float pos = ps[p];
      float aw  = at[p];
      int  i0 = (int)pos;                 // pos in [0,319] -> trunc == floor
      float fr = pos - (float)i0;
      int  i1 = min(i0 + 1, WWID - 1);
      v16h s0 = *(const v16h*)(vbuf + i0 * CC + g * 16);  // ds_load_b128 x2
      v16h s1 = *(const v16h*)(vbuf + i1 * CC + g * 16);
#pragma unroll
      for (int c = 0; c < 16; ++c) {
        float a0 = (float)s0[c], a1 = (float)s1[c];
        acc[c] += (a0 + (a1 - a0) * fr) * aw;
      }
    }
    int tile = w >> 4, m = w & 15;
#pragma unroll
    for (int c = 0; c < 16; ++c)
      aggbuf[tile * 2048 + a_frag_addr(m, g * 16 + c)] = (_Float16)acc[c];
  }
  __syncthreads();

  // ---- phase 3: upd = agg @ W_out + b_out ; x = cat1 + upd -> xrow ---------
  const float* xsrc = (b < 2) ? (f1 + b * CC * HW) : (f2 + (b - 2) * CC * HW);
  const int nn = lane & 15;
  const int mb = (lane >> 4) * 8;
  for (int t = wave; t < 20; t += 8) {
    v16h a[4];
#pragma unroll
    for (int s = 0; s < 4; ++s)
      a[s] = *(const v16h*)(aggbuf + t * 2048 + s * 512 + lane * 16);
#pragma unroll
    for (int nt = 0; nt < 8; ++nt) {
      v8f acc = {};
#pragma unroll
      for (int s = 0; s < 4; ++s) {
        v16h bf = *(const v16h*)(wb + OFF_WOUT + (nt * 4 + s) * 512 + lane * 16);
        acc = __builtin_amdgcn_wmma_f32_16x16x32_f16(false, a[s], false, bf,
                                                     (short)0, acc, false, false);
      }
      int n = nt * 16 + nn;
      float bias = b_out[n];
#pragma unroll
      for (int r = 0; r < 8; ++r) {
        int w = t * 16 + mb + r;
        xrow[w * CC + n] = acc[r] + bias + xsrc[n * HW + h * WWID + w];
      }
    }
  }
  __syncthreads();

  // ---- phase 4: LayerNorm over channels -> ws_xln (b128 traffic) -----------
  for (int w = threadIdx.x; w < WWID; w += 256) {
    const float4* xr4 = (const float4*)(xrow + w * CC);
    float mean = 0.0f;
#pragma unroll
    for (int i = 0; i < 32; ++i) {
      float4 t = xr4[i];
      mean += t.x + t.y + t.z + t.w;
    }
    mean *= (1.0f / 128.0f);
    float var = 0.0f;
#pragma unroll
    for (int i = 0; i < 32; ++i) {
      float4 t = xr4[i];
      float dx = t.x - mean, dy = t.y - mean, dz = t.z - mean, dw = t.w - mean;
      var += dx * dx + dy * dy + dz * dz + dw * dw;
    }
    var *= (1.0f / 128.0f);
    float inv = rsqrtf(var + 1e-5f);
    const float4* lw4 = (const float4*)ln_w;
    const float4* lb4 = (const float4*)ln_b;
    float4* dst4 = (float4*)(ws_xln + (row0 + w) * CC);
#pragma unroll
    for (int i = 0; i < 32; ++i) {
      float4 t = xr4[i], g = lw4[i], bb = lb4[i], o;
      o.x = (t.x - mean) * inv * g.x + bb.x;
      o.y = (t.y - mean) * inv * g.y + bb.y;
      o.z = (t.z - mean) * inv * g.z + bb.z;
      o.w = (t.w - mean) * inv * g.w + bb.w;
      dst4[i] = o;
    }
  }
}

// ---------------------------------------------------------------------------
// Kernel 3: FFN. h1 = silu(xln @ W1^T + b1); out = xln + h1 @ W2^T + b2.
// Per-wave 16-token tile; h1 kept in LDS in A-operand order (K=256, 8 steps).
// Transposed store through LDS into channel-planar f1o/f2o.
// ---------------------------------------------------------------------------
__global__ void __launch_bounds__(256) kffn(
    const float* __restrict__ ffn_b1, const float* __restrict__ ffn_b2,
    const _Float16* __restrict__ wb, const float* __restrict__ ws_xln,
    float* __restrict__ d_out)
{
  extern __shared__ char smem[];
  const int wave = threadIdx.x >> 5;
  const int lane = threadIdx.x & 31;
  _Float16* A1   = (_Float16*)smem + wave * 2048;            // xln f16, K=128
  _Float16* H1   = (_Float16*)(smem + 32768) + wave * 4096;  // silu out, K=256
  float*    OUTB = (float*)(smem + 98304) + wave * 2048;     // [c][m] f32

  const int tok0 = blockIdx.x * 128 + wave * 16;
  const int b    = tok0 / HW;
  const int rem  = tok0 % HW;
  const int h    = rem / WWID;
  const int w0   = rem % WWID;

  for (int i = lane; i < 2048; i += 32) {
    int c = i >> 4, m = i & 15;
    A1[a_frag_addr(m, c)] = (_Float16)ws_xln[(tok0 + m) * CC + c];
  }
  __syncthreads();

  v16h a[4];
#pragma unroll
  for (int s = 0; s < 4; ++s) a[s] = *(const v16h*)(A1 + s * 512 + lane * 16);

  const int nn = lane & 15;
  const int mb = (lane >> 4) * 8;
  // GEMM1 (N=256) + SiLU -> H1 (A-operand order for K=256)
#pragma unroll
  for (int nt = 0; nt < 16; ++nt) {
    v8f acc = {};
#pragma unroll
    for (int s = 0; s < 4; ++s) {
      v16h bf = *(const v16h*)(wb + OFF_W1 + (nt * 4 + s) * 512 + lane * 16);
      acc = __builtin_amdgcn_wmma_f32_16x16x32_f16(false, a[s], false, bf,
                                                   (short)0, acc, false, false);
    }
    int n = nt * 16 + nn;                 // 0..255 = K index of GEMM2
    float bias = ffn_b1[n];
    int s2 = n >> 5, kk = n & 31;
    int half = (kk & 15) >> 3;
    int e = ((kk >> 4) << 3) + (kk & 7);
#pragma unroll
    for (int r = 0; r < 8; ++r) {
      float hv = acc[r] + bias;
      hv = hv / (1.0f + __expf(-hv));     // SiLU
      H1[s2 * 512 + (half * 16 + (mb + r)) * 16 + e] = (_Float16)hv;
    }
  }
  __syncthreads();

  // GEMM2 (K=256, N=128) + residual -> OUTB[c][m]
#pragma unroll
  for (int nt = 0; nt < 8; ++nt) {
    v8f acc = {};
#pragma unroll
    for (int s = 0; s < 8; ++s) {
      v16h af = *(const v16h*)(H1 + s * 512 + lane * 16);
      v16h bf = *(const v16h*)(wb + OFF_W2 + (nt * 8 + s) * 512 + lane * 16);
      acc = __builtin_amdgcn_wmma_f32_16x16x32_f16(false, af, false, bf,
                                                   (short)0, acc, false, false);
    }
    int n = nt * 16 + nn;
    float bias = ffn_b2[n];
#pragma unroll
    for (int r = 0; r < 8; ++r) {
      int m = mb + r;
      OUTB[n * 16 + m] = ws_xln[(tok0 + m) * CC + n] + acc[r] + bias;
    }
  }
  __syncthreads();

  // transposed store: contiguous 16-token runs per channel plane
  for (int i = lane; i < 2048; i += 32) {
    int c = i >> 4, m = i & 15;
    float val = OUTB[c * 16 + m];
    int o = (b < 2) ? ((b * CC + c) * HHH + h) * WWID + w0 + m
                    : FOUT + (((b - 2) * CC + c) * HHH + h) * WWID + w0 + m;
    d_out[o] = val;
  }
}

// ---------------------------------------------------------------------------
extern "C" void kernel_launch(void* const* d_in, const int* in_sizes, int n_in,
                              void* d_out, int out_size, void* d_ws, size_t ws_size,
                              hipStream_t stream) {
  (void)in_sizes; (void)n_in; (void)out_size; (void)ws_size;
  const float* f1     = (const float*)d_in[0];
  const float* f2     = (const float*)d_in[1];
  const float* anchor = (const float*)d_in[2];
  const float* Wv     = (const float*)d_in[3];
  const float* bv     = (const float*)d_in[4];
  const float* Woff   = (const float*)d_in[5];
  const float* boff   = (const float*)d_in[6];
  const float* Wattn  = (const float*)d_in[7];
  const float* battn  = (const float*)d_in[8];
  const float* Wout   = (const float*)d_in[9];
  const float* bout   = (const float*)d_in[10];
  const float* lnw    = (const float*)d_in[11];
  const float* lnb    = (const float*)d_in[12];
  const float* W1     = (const float*)d_in[13];
  const float* b1     = (const float*)d_in[14];
  const float* W2     = (const float*)d_in[15];
  const float* b2     = (const float*)d_in[16];
  float* out = (float*)d_out;
  char*  ws  = (char*)d_ws;
  _Float16* wb     = (_Float16*)ws;
  _Float16* ws_v   = (_Float16*)(ws + WSB_V);
  float*    ws_at  = (float*)(ws + WSB_ATTN);
  float*    ws_ps  = (float*)(ws + WSB_POS);
  float*    ws_xln = (float*)(ws + WSB_XLN);

  hipLaunchKernelGGL(kprep,   dim3(416), dim3(256), 0,      stream,
                     Wv, Woff, Wattn, Wout, W1, W2, wb);
  hipLaunchKernelGGL(kproj,   dim3(NTOK / 128), dim3(256), 98304, stream,
                     f1, f2, anchor, bv, boff, battn, wb, ws_v, ws_at, ws_ps, out);
  hipLaunchKernelGGL(ksample, dim3(BBB * HHH), dim3(256), 245760, stream,
                     f1, f2, bout, lnw, lnb, wb, ws_v, ws_at, ws_ps, ws_xln);
  hipLaunchKernelGGL(kffn,    dim3(NTOK / 128), dim3(256), 163840, stream,
                     b1, b2, wb, ws_xln, out);
}
